// CapsuleNet_14053132993109
// MI455X (gfx1250) — matrix operand order repaired and checked by reading, not compiled, not run on hardware
//
#include <hip/hip_runtime.h>
#include <cstdint>
#include <cstddef>

typedef _Float16 half_t;
typedef __attribute__((ext_vector_type(16))) _Float16 v16h;
typedef __attribute__((ext_vector_type(8)))  _Float16 v8h;
typedef __attribute__((ext_vector_type(8)))  float    v8f;
typedef __attribute__((ext_vector_type(2)))  float    v2f;

#define BATCH   128
#define NMAP    32
#define NPRIM   144
#define NIN     4608   // 32*144
#define NCLS    10
#define ODIM    16
#define IDIM    8

// scheduling-directive helper: (mask, count) groups; 0x100 = DS read, 0x8 = WMMA
#if defined(__has_builtin) && __has_builtin(__builtin_amdgcn_sched_group_barrier)
#define SGB(mask, n) __builtin_amdgcn_sched_group_barrier((mask), (n), 0)
#else
#define SGB(mask, n)
#endif

// ---------------------------------------------------------------------------
// conv1 (1->256ch, 9x9, 28x28 -> 20x20) + ReLU, output NHWC f16 x[b][h][w][co]
// ---------------------------------------------------------------------------
__global__ __launch_bounds__(256) void conv1_relu_kernel(
    const float* __restrict__ img, const float* __restrict__ w,
    const float* __restrict__ bias, half_t* __restrict__ xh)
{
    __shared__ float simg[784];
    __shared__ float sw[16 * 81];
    __shared__ float sb[16];
    const int b  = blockIdx.x;
    const int cg = blockIdx.y;
    for (int i = threadIdx.x; i < 784; i += 256) simg[i] = img[b * 784 + i];
    for (int i = threadIdx.x; i < 16 * 81; i += 256) sw[i] = w[cg * 16 * 81 + i];
    if (threadIdx.x < 16) sb[threadIdx.x] = bias[cg * 16 + threadIdx.x];
    __syncthreads();

    for (int o = threadIdx.x; o < 16 * 400; o += 256) {
        const int cl  = o / 400;
        const int pix = o - cl * 400;
        const int oh  = pix / 20, ow = pix - oh * 20;
        float s = sb[cl];
        const float* wp = sw + cl * 81;
        #pragma unroll
        for (int kh = 0; kh < 9; ++kh)
            #pragma unroll
            for (int kw = 0; kw < 9; ++kw)
                s += simg[(oh + kh) * 28 + (ow + kw)] * wp[kh * 9 + kw];
        s = fmaxf(s, 0.f);
        const int co = cg * 16 + cl;
        xh[(((size_t)b * 20 + oh) * 20 + ow) * 256 + co] = (half_t)s;
    }
}

// ---------------------------------------------------------------------------
// conv2 weight relayout: f32 [co][ci][9][9] -> f16 [r=kh*9+kw][co][ci]
// ---------------------------------------------------------------------------
__global__ void cast_w2_kernel(const float* __restrict__ src, half_t* __restrict__ dst)
{
    const int id = blockIdx.x * 256 + threadIdx.x;
    if (id >= 81 * 256 * 256) return;
    const int ci = id & 255;
    const int t  = id >> 8;
    const int co = t & 255;
    const int r  = t >> 8;
    dst[id] = (half_t)src[(size_t)(co * 256 + ci) * 81 + r];
}

// ---------------------------------------------------------------------------
// conv2 (256->256, 9x9, 20x20 -> 12x12) as 81 shifted GEMMs over ci, WMMA f16.
// grid (128 batches, 2 co-halves), 8 waves/block, wave = one 16-row co tile.
//
// B slice per (kh,kw,ks) step (144 pix x 32 ci = 9 KB) is shared by all 8
// waves -> double-buffered LDS staging by threads 0..191 (wave-uniform guard,
// exactly 3 x 16B quarters each). Software pipelining:
//   - A fragment double-buffered across steps (a_cur / a_nxt)
//   - 9-WMMA burst with DS-read/WMMA interleave enforced by
//     sched_group_barrier: [DS*4][W][DS*2][W]x7[W][W]  (2-deep pipeline)
//   - staging global loads issued before the WMMA burst, ds_stores after it
// ---------------------------------------------------------------------------
__global__ __launch_bounds__(256) void conv2_wmma_kernel(
    const half_t* __restrict__ xh, const half_t* __restrict__ w2h,
    const float* __restrict__ c2b, float* __restrict__ y2)
{
    __shared__ __align__(64) half_t sbuf[2][144 * 32];   // 2 x 9 KB
    const int tid   = threadIdx.x;
    const int b     = blockIdx.x;
    const int mhalf = blockIdx.y;
    const int wave  = tid >> 5;
    const int lane  = tid & 31;
    const int l15   = lane & 15;
    const int half  = lane >> 4;
    const int mtile = mhalf * 8 + wave;          // 0..15
    const int mrow  = mtile * 16 + l15;          // A-row (out channel)

    const half_t* xb    = xh + (size_t)b * (20 * 20 * 256);
    const half_t* abase = w2h + (size_t)mrow * 256 + half * 16;  // + r*65536 + ks*32

    // staging: threads 0..191 move exactly 3 x 16B quarters (576 = 192*3)
    const bool doStage = (tid < 192);
    int slotPO[3], slotOfs[3], slotQ8[3];
    #pragma unroll
    for (int j = 0; j < 3; ++j) {
        const int idx = tid * 3 + j;             // 0..575 when doStage
        const int pix = (idx >> 2) & 255;        // 0..143
        const int q   = idx & 3;
        const int oh  = pix / 12, ow = pix - oh * 12;
        slotPO[j]  = oh * 20 + ow;               // pixel offset in 20x20 plane
        slotQ8[j]  = q * 8;                      // quarter offset (halves)
        slotOfs[j] = pix * 32 + q * 8;           // LDS offset (halves)
    }

    v8f acc[9];
    #pragma unroll
    for (int t = 0; t < 9; ++t) { v8f z = {}; acc[t] = z; }

    // 9-WMMA burst; sched_group_barrier pins a 2-deep DS-read/WMMA pipeline
    auto burst = [&](const half_t* bbase, const v16h& a) {
        v16h b0 = *(const v16h*)(bbase);
        v16h b1 = *(const v16h*)(bbase + 512);
        acc[0] = __builtin_amdgcn_wmma_f32_16x16x32_f16(false, a, false, b0, (short)0, acc[0], false, false);
        b0 = *(const v16h*)(bbase + 2 * 512);
        acc[1] = __builtin_amdgcn_wmma_f32_16x16x32_f16(false, a, false, b1, (short)0, acc[1], false, false);
        b1 = *(const v16h*)(bbase + 3 * 512);
        acc[2] = __builtin_amdgcn_wmma_f32_16x16x32_f16(false, a, false, b0, (short)0, acc[2], false, false);
        b0 = *(const v16h*)(bbase + 4 * 512);
        acc[3] = __builtin_amdgcn_wmma_f32_16x16x32_f16(false, a, false, b1, (short)0, acc[3], false, false);
        b1 = *(const v16h*)(bbase + 5 * 512);
        acc[4] = __builtin_amdgcn_wmma_f32_16x16x32_f16(false, a, false, b0, (short)0, acc[4], false, false);
        b0 = *(const v16h*)(bbase + 6 * 512);
        acc[5] = __builtin_amdgcn_wmma_f32_16x16x32_f16(false, a, false, b1, (short)0, acc[5], false, false);
        b1 = *(const v16h*)(bbase + 7 * 512);
        acc[6] = __builtin_amdgcn_wmma_f32_16x16x32_f16(false, a, false, b0, (short)0, acc[6], false, false);
        b0 = *(const v16h*)(bbase + 8 * 512);
        acc[7] = __builtin_amdgcn_wmma_f32_16x16x32_f16(false, a, false, b1, (short)0, acc[7], false, false);
        acc[8] = __builtin_amdgcn_wmma_f32_16x16x32_f16(false, a, false, b0, (short)0, acc[8], false, false);
        // desired schedule: 2 DS pairs ahead of each WMMA
        SGB(0x100, 4);   // DS reads: pairs 0,1
        SGB(0x008, 1);   // WMMA 0
        #pragma unroll
        for (int g = 0; g < 7; ++g) {
            SGB(0x100, 2);   // next DS pair
            SGB(0x008, 1);   // WMMA g+1
        }
        SGB(0x008, 1);   // WMMA 8
    };

    // stage step 0 into buffer 0   (kh=kw=0, ks=0)
    if (doStage) {
        #pragma unroll
        for (int j = 0; j < 3; ++j) {
            const v8h s0 = *(const v8h*)(xb + (size_t)slotPO[j] * 256 + slotQ8[j]);
            *(v8h*)(&sbuf[0][slotOfs[j]]) = s0;
        }
    }
    __syncthreads();

    // preload A fragment for step 0
    v16h a_cur;
    {
        const v8h alo = *(const v8h*)abase;
        const v8h ahi = *(const v8h*)(abase + 8);
        #pragma unroll
        for (int j = 0; j < 8; ++j) { a_cur[j] = alo[j]; a_cur[8 + j] = ahi[j]; }
    }

    for (int step = 0; step < 647; ++step) {     // 81 (kh,kw) x 8 ci-chunks
        const int cur = step & 1;

        // 1) issue global loads for next step's B slice + next A fragment
        const int s1  = step + 1;
        const int r1  = s1 >> 3, ks1 = s1 & 7;
        const int kh1 = r1 / 9,  kw1 = r1 - kh1 * 9;
        const size_t gbase = (size_t)(kh1 * 20 + kw1) * 256 + ks1 * 32;
        v8h sreg[3];
        if (doStage) {
            #pragma unroll
            for (int j = 0; j < 3; ++j)
                sreg[j] = *(const v8h*)(xb + (size_t)slotPO[j] * 256 + gbase + slotQ8[j]);
        }
        v16h a_nxt;
        {
            const half_t* aptr = abase + (size_t)r1 * 65536 + ks1 * 32;
            const v8h alo = *(const v8h*)aptr;
            const v8h ahi = *(const v8h*)(aptr + 8);
            #pragma unroll
            for (int j = 0; j < 8; ++j) { a_nxt[j] = alo[j]; a_nxt[8 + j] = ahi[j]; }
        }

        // 2) 9-WMMA burst from LDS
        burst(&sbuf[cur][l15 * 32 + half * 16], a_cur);

        // 3) commit staged slice to the other buffer, rotate A, sync
        if (doStage) {
            #pragma unroll
            for (int j = 0; j < 3; ++j)
                *(v8h*)(&sbuf[cur ^ 1][slotOfs[j]]) = sreg[j];
        }
        a_cur = a_nxt;
        __syncthreads();
    }

    // final step (647): compute only
    burst(&sbuf[1][l15 * 32 + half * 16], a_cur);

    // store y2[b][co][pixel] + bias
    #pragma unroll
    for (int t = 0; t < 9; ++t) {
        const int col = t * 16 + l15;
        #pragma unroll
        for (int vr = 0; vr < 8; ++vr) {
            const int co = mtile * 16 + half * 8 + vr;
            y2[((size_t)b * 256 + co) * 144 + col] = acc[t][vr] + c2b[co];
        }
    }
}

// ---------------------------------------------------------------------------
// squash primary capsules: y2[b][co=i*32+m][p] -> u[b][m*144+p][i]  (f32)
// ---------------------------------------------------------------------------
__global__ void squash_u_kernel(const float* __restrict__ y2, float* __restrict__ u)
{
    const int id = blockIdx.x * 256 + threadIdx.x;
    if (id >= BATCH * NIN) return;
    const int b  = id / NIN;
    const int mp = id - b * NIN;
    const int m  = mp / NPRIM;
    const int p  = mp - m * NPRIM;
    float vals[IDIM];
    float s2 = 0.f;
    #pragma unroll
    for (int i = 0; i < IDIM; ++i) {
        const float x = y2[((size_t)b * 256 + (i * 32 + m)) * 144 + p];
        vals[i] = x;
        s2 += x * x;
    }
    const float sc = (s2 / (1.f + s2)) / sqrtf(s2 + 1e-8f);
    #pragma unroll
    for (int i = 0; i < IDIM; ++i)
        u[(size_t)id * IDIM + i] = vals[i] * sc;
}

// ---------------------------------------------------------------------------
// u_hat[b,n,c,o] = sum_i W[n,c,o,i] * u[b,n,i]  via f32 WMMA 16x16x4 (K=8=2x4)
// Output stored f16 (189 MB -> fits 192 MB L2 for the routing sweeps).
// ---------------------------------------------------------------------------
__global__ __launch_bounds__(256) void uhat_wmma_kernel(
    const float* __restrict__ W, const float* __restrict__ u, half_t* __restrict__ uh)
{
    const int mp   = blockIdx.x;            // 0..4607
    const int wave = threadIdx.x >> 5;      // batch tile 0..7
    const int lane = threadIdx.x & 31;
    const int l15  = lane & 15;
    const int half = lane >> 4;

    const float* Wmp  = W + (size_t)mp * (NCLS * ODIM) * IDIM;
    const int    bcol = wave * 16 + l15;                    // batch column
    const float* ub   = u + ((size_t)bcol * NIN + mp) * IDIM;

    #pragma unroll
    for (int mt = 0; mt < 10; ++mt) {
        v8f acc = {};
        const float* arow = Wmp + (size_t)(mt * 16 + l15) * IDIM;
        #pragma unroll
        for (int ks = 0; ks < 2; ++ks) {
            const int kb = ks * 4 + half * 2;
            const v2f a  = *(const v2f*)(arow + kb);
            const v2f bf = *(const v2f*)(ub + kb);
            acc = __builtin_amdgcn_wmma_f32_16x16x4_f32(
                false, a, false, bf, (short)0, acc, false, false);
        }
        #pragma unroll
        for (int vr = 0; vr < 8; ++vr) {
            const int coi = mt * 16 + half * 8 + vr;        // c*16+o
            uh[((size_t)bcol * NIN + mp) * 160 + coi] = (half_t)acc[vr];
        }
    }
}

// ---------------------------------------------------------------------------
// routing helpers
// ---------------------------------------------------------------------------
__global__ void zero_f32_kernel(float* __restrict__ p, int n)
{
    const int id = blockIdx.x * 256 + threadIdx.x;
    if (id < n) p[id] = 0.f;
}

// per (b,c): softmax over n of b_ij, s_j = sum_n c*u_hat, v = squash(s_j)
__global__ __launch_bounds__(256) void route_kernel(
    const half_t* __restrict__ uh, const float* __restrict__ bij, float* __restrict__ v)
{
    __shared__ float red[256];
    __shared__ float svec[16];
    __shared__ float msh, zsh;
    const int bc  = blockIdx.x;       // b*10 + c
    const int b   = bc / NCLS;
    const int c   = bc - b * NCLS;
    const int tid = threadIdx.x;
    const float* brow = bij + (size_t)bc * NIN;

    float m = -3.4e38f;
    for (int n = tid; n < NIN; n += 256) m = fmaxf(m, brow[n]);
    red[tid] = m; __syncthreads();
    for (int s = 128; s > 0; s >>= 1) {
        if (tid < s) red[tid] = fmaxf(red[tid], red[tid + s]);
        __syncthreads();
    }
    if (tid == 0) msh = red[0];
    __syncthreads();

    float z = 0.f;
    float sp[ODIM];
    #pragma unroll
    for (int d = 0; d < ODIM; ++d) sp[d] = 0.f;
    for (int n = tid; n < NIN; n += 256) {
        const float w = __expf(brow[n] - msh);
        z += w;
        const v16h uv = *(const v16h*)(uh + (((size_t)b * NIN + n) * NCLS + c) * ODIM);
        #pragma unroll
        for (int d = 0; d < ODIM; ++d) sp[d] += w * (float)uv[d];
    }
    red[tid] = z; __syncthreads();
    for (int s = 128; s > 0; s >>= 1) {
        if (tid < s) red[tid] += red[tid + s];
        __syncthreads();
    }
    if (tid == 0) zsh = red[0];
    if (tid < 16) svec[tid] = 0.f;
    __syncthreads();
    #pragma unroll
    for (int d = 0; d < ODIM; ++d) atomicAdd(&svec[d], sp[d]);
    __syncthreads();

    if (tid == 0) {
        const float invz = 1.f / zsh;
        float sv[ODIM];
        float s2 = 0.f;
        #pragma unroll
        for (int d = 0; d < ODIM; ++d) { sv[d] = svec[d] * invz; s2 += sv[d] * sv[d]; }
        const float sc = (s2 / (1.f + s2)) / sqrtf(s2 + 1e-8f);
        #pragma unroll
        for (int d = 0; d < ODIM; ++d) v[(size_t)bc * ODIM + d] = sv[d] * sc;
    }
}

// agreement update: b_ij[b,c,n] += dot(u_hat[b,n,c,:], v[b,c,:])
__global__ __launch_bounds__(256) void agree_kernel(
    const half_t* __restrict__ uh, const float* __restrict__ v, float* __restrict__ bij)
{
    __shared__ float vloc[ODIM];
    const int bc = blockIdx.x;
    const int b  = bc / NCLS;
    const int c  = bc - b * NCLS;
    if (threadIdx.x < ODIM) vloc[threadIdx.x] = v[(size_t)bc * ODIM + threadIdx.x];
    __syncthreads();
    for (int n = threadIdx.x; n < NIN; n += 256) {
        const v16h uv = *(const v16h*)(uh + (((size_t)b * NIN + n) * NCLS + c) * ODIM);
        float dot = 0.f;
        #pragma unroll
        for (int d = 0; d < ODIM; ++d) dot += (float)uv[d] * vloc[d];
        bij[(size_t)bc * NIN + n] += dot;
    }
}

// ---------------------------------------------------------------------------
// v_length, argmax -> one-hot (floats), masked capsule vector (f16)
// ---------------------------------------------------------------------------
__global__ void mask_argmax_kernel(const float* __restrict__ v,
                                   float* __restrict__ out, half_t* __restrict__ masked)
{
    const int b = threadIdx.x;
    if (b >= BATCH) return;
    float len[NCLS];
    float best = -1.f;
    int   bi   = 0;
    for (int c = 0; c < NCLS; ++c) {
        float s = 0.f;
        #pragma unroll
        for (int d = 0; d < ODIM; ++d) {
            const float x = v[((size_t)b * NCLS + c) * ODIM + d];
            s += x * x;
        }
        len[c] = sqrtf(s);
        if (len[c] > best) { best = len[c]; bi = c; }
    }
    for (int c = 0; c < NCLS; ++c) {
        out[b * NCLS + c] = (c == bi) ? 1.f : 0.f;                 // y_pred_ohe
        out[1280 + 100352 + b * NCLS + c] = len[c];                // v_length
        #pragma unroll
        for (int d = 0; d < ODIM; ++d)
            masked[(size_t)b * 160 + c * ODIM + d] =
                (c == bi) ? (half_t)v[((size_t)b * NCLS + c) * ODIM + d] : (half_t)0.0f;
    }
}

// ---------------------------------------------------------------------------
// decoder weight transpose-cast: f32 [K][N] -> f16 [N][K]
// ---------------------------------------------------------------------------
__global__ void tcast_kernel(const float* __restrict__ src, half_t* __restrict__ dst,
                             int K, int N)
{
    const int id = blockIdx.x * 256 + threadIdx.x;
    if (id >= K * N) return;
    const int k = id % K;
    const int n = id / K;
    dst[id] = (half_t)src[(size_t)k * N + n];
}

// ---------------------------------------------------------------------------
// decoder GEMM: out[128 x N] = act(A[128 x K] @ Bt[N x K]^T + bias), WMMA f16.
// act 0 = ReLU -> f16 out, act 1 = sigmoid -> f32 out
// ---------------------------------------------------------------------------
__global__ __launch_bounds__(256) void dec_gemm_kernel(
    const half_t* __restrict__ A, const half_t* __restrict__ Bt,
    const float* __restrict__ bias, half_t* __restrict__ outh,
    float* __restrict__ outf, int K, int N, int act)
{
    const int ntile = blockIdx.x;
    const int wave  = threadIdx.x >> 5;
    const int lane  = threadIdx.x & 31;
    const int l15   = lane & 15;
    const int half  = lane >> 4;

    const half_t* arow = A  + (size_t)(wave * 16 + l15) * K + half * 16;
    const half_t* bcol = Bt + (size_t)(ntile * 16 + l15) * K + half * 16;

    v8f acc = {};
    for (int ks = 0; ks < (K >> 5); ++ks) {
        const int ko = ks * 32;
        const v8h alo = *(const v8h*)(arow + ko);
        const v8h ahi = *(const v8h*)(arow + ko + 8);
        v16h a;
        #pragma unroll
        for (int j = 0; j < 8; ++j) { a[j] = alo[j]; a[8 + j] = ahi[j]; }
        const v16h bb = *(const v16h*)(bcol + ko);
        acc = __builtin_amdgcn_wmma_f32_16x16x32_f16(
            false, a, false, bb, (short)0, acc, false, false);
    }
    const int col = ntile * 16 + l15;
    const float bv = bias[col];
    #pragma unroll
    for (int vr = 0; vr < 8; ++vr) {
        const int row = wave * 16 + half * 8 + vr;
        const float val = acc[vr] + bv;
        if (act == 0) outh[(size_t)row * N + col] = (half_t)fmaxf(val, 0.f);
        else          outf[(size_t)row * N + col] = 1.f / (1.f + __expf(-val));
    }
}

// ---------------------------------------------------------------------------
// host-side launch
// ---------------------------------------------------------------------------
extern "C" void kernel_launch(void* const* d_in, const int* in_sizes, int n_in,
                              void* d_out, int out_size, void* d_ws, size_t ws_size,
                              hipStream_t stream)
{
    (void)in_sizes; (void)n_in; (void)out_size; (void)ws_size;
    const float* imgs = (const float*)d_in[0];
    const float* c1w  = (const float*)d_in[1];
    const float* c1b  = (const float*)d_in[2];
    const float* c2w  = (const float*)d_in[3];
    const float* c2b  = (const float*)d_in[4];
    const float* Wd   = (const float*)d_in[5];
    const float* dw1  = (const float*)d_in[6];
    const float* db1  = (const float*)d_in[7];
    const float* dw2  = (const float*)d_in[8];
    const float* db2  = (const float*)d_in[9];
    const float* dw3  = (const float*)d_in[10];
    const float* db3  = (const float*)d_in[11];
    float* out = (float*)d_out;

    char*  ws  = (char*)d_ws;
    size_t off = 0;
    auto alloc = [&](size_t bytes) -> char* {
        char* p = ws + off;
        off += (bytes + 255) & ~(size_t)255;
        return p;
    };

    half_t* xh     = (half_t*)alloc((size_t)BATCH * 400 * 256 * 2);        // 26.2 MB
    half_t* w2h    = (half_t*)alloc((size_t)81 * 256 * 256 * 2);           // 10.6 MB
    float*  y2     = (float*) alloc((size_t)BATCH * 256 * 144 * 4);        // 18.9 MB
    float*  u      = (float*) alloc((size_t)BATCH * NIN * IDIM * 4);       // 18.9 MB
    half_t* uh     = (half_t*)alloc((size_t)BATCH * NIN * 160 * 2);        // 188.7 MB (L2-resident)
    float*  bij    = (float*) alloc((size_t)BATCH * NCLS * NIN * 4);       // 23.6 MB
    float*  vbuf   = (float*) alloc((size_t)BATCH * NCLS * ODIM * 4);
    half_t* masked = (half_t*)alloc((size_t)BATCH * 160 * 2);
    half_t* w1t    = (half_t*)alloc((size_t)512 * 160 * 2);
    half_t* w2t    = (half_t*)alloc((size_t)1024 * 512 * 2);
    half_t* w3t    = (half_t*)alloc((size_t)784 * 1024 * 2);
    half_t* h1     = (half_t*)alloc((size_t)BATCH * 512 * 2);
    half_t* h2     = (half_t*)alloc((size_t)BATCH * 1024 * 2);

    // conv1 + relu (NHWC f16)
    conv1_relu_kernel<<<dim3(BATCH, 16), 256, 0, stream>>>(imgs, c1w, c1b, xh);
    // conv2 weight relayout
    cast_w2_kernel<<<(81 * 256 * 256 + 255) / 256, 256, 0, stream>>>(c2w, w2h);
    // conv2 via WMMA f16 (dominant 196 GFLOP)
    conv2_wmma_kernel<<<dim3(BATCH, 2), 256, 0, stream>>>(xh, w2h, c2b, y2);
    // primary capsule squash
    squash_u_kernel<<<(BATCH * NIN + 255) / 256, 256, 0, stream>>>(y2, u);
    // u_hat via WMMA f32 16x16x4, stored f16
    uhat_wmma_kernel<<<NIN, 256, 0, stream>>>(Wd, u, uh);

    // dynamic routing (3 iterations)
    const int nbij = BATCH * NCLS * NIN;
    zero_f32_kernel<<<(nbij + 255) / 256, 256, 0, stream>>>(bij, nbij);
    for (int it = 0; it < 3; ++it) {
        route_kernel<<<BATCH * NCLS, 256, 0, stream>>>(uh, bij, vbuf);
        if (it < 2)
            agree_kernel<<<BATCH * NCLS, 256, 0, stream>>>(uh, vbuf, bij);
    }

    // outputs 0 and 2 + masked decoder input
    mask_argmax_kernel<<<1, BATCH, 0, stream>>>(vbuf, out, masked);

    // decoder weight transpose-casts
    tcast_kernel<<<(160 * 512 + 255) / 256, 256, 0, stream>>>(dw1, w1t, 160, 512);
    tcast_kernel<<<(512 * 1024 + 255) / 256, 256, 0, stream>>>(dw2, w2t, 512, 1024);
    tcast_kernel<<<(1024 * 784 + 255) / 256, 256, 0, stream>>>(dw3, w3t, 1024, 784);

    // decoder GEMMs via WMMA f16
    dec_gemm_kernel<<<512 / 16, 256, 0, stream>>>(masked, w1t, db1, h1, nullptr, 160, 512, 0);
    dec_gemm_kernel<<<1024 / 16, 256, 0, stream>>>(h1, w2t, db2, h2, nullptr, 512, 1024, 0);
    dec_gemm_kernel<<<784 / 16, 256, 0, stream>>>(h2, w3t, db3, nullptr, out + 1280, 1024, 784, 1);
}
